// EdgeWeightPredictorMLP_68624987455985
// MI455X (gfx1250) — matrix operand order repaired and checked by reading, not compiled, not run on hardware
//
#include <hip/hip_runtime.h>
#include <stdint.h>

// ---------------------------------------------------------------------------
// EdgeWeightPredictorMLP for MI455X (gfx1250, wave32, WMMA 16x16x32 bf16)
//
//  L1: h1 = ef @ W1^T + b1          ef = [x[src] | x[dst]]  (K=256)
//  L2: h2 = relu(bn1(h1)) @ W2^T+b2 (K=128)
//  L3: out = relu(bn2(h2)) . w3 + b3
//
//  BN(training) folded into per-channel affine (A,B) computed by a
//  deterministic two-stage reduction (no atomics).
// ---------------------------------------------------------------------------

typedef __attribute__((ext_vector_type(16))) __bf16 v16bf;
typedef __attribute__((ext_vector_type(2)))  __bf16 v2bf;
typedef __attribute__((ext_vector_type(8)))  float  v8f;
typedef __attribute__((ext_vector_type(2)))  float  v2f;

struct Frag8 { uint32_t u[8]; };   // one 16x32 / 32x16 bf16 WMMA fragment (8 VGPRs)

#define NBLK   2048   // persistent blocks for the two GEMM kernels
#define WAVES  4      // waves per block (128 threads)

__device__ __forceinline__ uint32_t pack2(float lo, float hi) {
  v2f f; f.x = lo; f.y = hi;
  v2bf b = __builtin_convertvector(f, v2bf);   // hw packed f32->bf16 cvt
  return __builtin_bit_cast(uint32_t, b);
}
__device__ __forceinline__ float bf2f(uint32_t h) {
  return __builtin_bit_cast(float, h << 16);
}

// ===========================================================================
// Kernel 1: layer-1 GEMM (gather + concat + [E,256]x[256,128]) + stats
// ===========================================================================
__global__ __launch_bounds__(128) void k_layer1(
    const float* __restrict__ x, const int* __restrict__ ei,
    const float* __restrict__ W1, const float* __restrict__ b1,
    uint16_t* __restrict__ h1, float* __restrict__ partial,
    int E, int ntiles)
{
  __shared__ __align__(16) uint32_t sW[128 * 132];        // W1 bf16, pitch 264 half = 132 dw
  __shared__ __align__(16) float    sStage[WAVES][16 * 132];
  __shared__ float sStats[WAVES][256];
  __shared__ float sBias[128];

  const int tid = threadIdx.x;

  // --- stage W1 (fp32 [128][256]) as bf16 into LDS, one row per thread ----
  {
    const int row = tid;
    const float4* src = reinterpret_cast<const float4*>(W1 + row * 256);
    uint32_t* drow = &sW[row * 132];
    #pragma unroll
    for (int k4 = 0; k4 < 64; ++k4) {
      float4 w = src[k4];
      drow[k4 * 2 + 0] = pack2(w.x, w.y);
      drow[k4 * 2 + 1] = pack2(w.z, w.w);
    }
    sBias[row] = b1[row];
  }
  __syncthreads();

  const int wave = tid >> 5, lane = tid & 31;
  const int mrow = lane & 15, half = lane >> 4;
  const int gwave  = blockIdx.x * WAVES + wave;
  const int nwaves = gridDim.x * WAVES;
  const int iters  = (ntiles + nwaves - 1) / nwaves;   // uniform trip count

  float biasr[8];
  #pragma unroll
  for (int nt = 0; nt < 8; ++nt) biasr[nt] = sBias[nt * 16 + mrow];

  float sumA[8], sumQ[8];
  #pragma unroll
  for (int i = 0; i < 8; ++i) { sumA[i] = 0.f; sumQ[i] = 0.f; }

  for (int it = 0; it < iters; ++it) {
    const int tile = gwave + it * nwaves;
    const bool tvalid = tile < ntiles;
    const int e0 = tile * 16;
    const bool fulltile = (e0 + 16 <= E);

    Frag8 af[8];
    if (tvalid) {
      int e = e0 + mrow;
      if (!fulltile && e >= E) e = E - 1;
      const float* ps = x + (size_t)ei[e] * 128;       // src features (k<128)
      const float* pd = x + (size_t)ei[E + e] * 128;   // dst features (k>=128)
      #pragma unroll
      for (int ks = 0; ks < 8; ++ks) {
        #pragma unroll
        for (int j = 0; j < 2; ++j) {
          int c = ks * 32 + half * 16 + j * 8;
          const float* p = (c < 128) ? (ps + c) : (pd + (c - 128));
          float4 a0 = *reinterpret_cast<const float4*>(p);
          float4 a1 = *reinterpret_cast<const float4*>(p + 4);
          af[ks].u[j * 4 + 0] = pack2(a0.x, a0.y);
          af[ks].u[j * 4 + 1] = pack2(a0.z, a0.w);
          af[ks].u[j * 4 + 2] = pack2(a1.x, a1.y);
          af[ks].u[j * 4 + 3] = pack2(a1.z, a1.w);
        }
      }
      #pragma unroll
      for (int nt = 0; nt < 8; ++nt) {
        const int brow = nt * 16 + mrow;
        const uint32_t* wbase = &sW[brow * 132 + half * 8];
        // ---- phase 1: issue all 16 ds_load_b128 (latency overlapped) ----
        Frag8 bfr[8];
        #pragma unroll
        for (int ks = 0; ks < 8; ++ks) {
          const uint4* bp = reinterpret_cast<const uint4*>(wbase + ks * 16);
          uint4 q0 = bp[0], q1 = bp[1];
          bfr[ks].u[0] = q0.x; bfr[ks].u[1] = q0.y;
          bfr[ks].u[2] = q0.z; bfr[ks].u[3] = q0.w;
          bfr[ks].u[4] = q1.x; bfr[ks].u[5] = q1.y;
          bfr[ks].u[6] = q1.z; bfr[ks].u[7] = q1.w;
        }
        // ---- phase 2: 8 chained WMMAs, bias pre-seeded in C -------------
        v8f acc;
        #pragma unroll
        for (int i = 0; i < 8; ++i) acc[i] = biasr[nt];
        #pragma unroll
        for (int ks = 0; ks < 8; ++ks) {
          acc = __builtin_amdgcn_wmma_f32_16x16x32_bf16(
              false, __builtin_bit_cast(v16bf, af[ks]),
              false, __builtin_bit_cast(v16bf, bfr[ks]),
              (short)0, acc, false, false);
        }
        const int n = nt * 16 + mrow;
        if (fulltile) {
          #pragma unroll
          for (int r = 0; r < 8; ++r) {
            float v = acc[r];
            sumA[nt] += v; sumQ[nt] += v * v;
            sStage[wave][(r + half * 8) * 132 + n] = v;
          }
        } else {
          #pragma unroll
          for (int r = 0; r < 8; ++r) {
            float v = acc[r];
            int mr = r + half * 8;
            if (e0 + mr < E) { sumA[nt] += v; sumQ[nt] += v * v; }
            sStage[wave][mr * 132 + n] = v;
          }
        }
      }
    }
    __syncthreads();
    if (tvalid) {     // coalesced bf16 store of the 16x128 tile
      #pragma unroll
      for (int i = 0; i < 8; ++i) {
        int row  = i * 2 + half;
        int colf = mrow * 8;
        if (fulltile || (e0 + row < E)) {
          const float* sp = &sStage[wave][row * 132 + colf];
          float4 f0 = *reinterpret_cast<const float4*>(sp);
          float4 f1 = *reinterpret_cast<const float4*>(sp + 4);
          uint4 o;
          o.x = pack2(f0.x, f0.y); o.y = pack2(f0.z, f0.w);
          o.z = pack2(f1.x, f1.y); o.w = pack2(f1.z, f1.w);
          size_t off = (size_t)(e0 + row) * 128 + colf;     // bf16 elements
          reinterpret_cast<uint4*>(h1)[off >> 3] = o;
        }
      }
    }
    __syncthreads();
  }

  // --- deterministic per-block partial stats -------------------------------
  #pragma unroll
  for (int nt = 0; nt < 8; ++nt) {
    float a = sumA[nt] + __shfl_xor(sumA[nt], 16);
    float q = sumQ[nt] + __shfl_xor(sumQ[nt], 16);
    if (half == 0) {
      sStats[wave][nt * 16 + mrow]       = a;
      sStats[wave][128 + nt * 16 + mrow] = q;
    }
  }
  __syncthreads();
  {
    const int t = tid;
    float s = sStats[0][t] + sStats[1][t] + sStats[2][t] + sStats[3][t];
    float q = sStats[0][128 + t] + sStats[1][128 + t] +
              sStats[2][128 + t] + sStats[3][128 + t];
    partial[(size_t)blockIdx.x * 256 + t]       = s;
    partial[(size_t)blockIdx.x * 256 + 128 + t] = q;
  }
}

// ===========================================================================
// Kernel: reduce per-block stats -> BN affine coeffs  A=g*rstd, B=be-mean*A
// grid = 128 (one block per channel), 256 threads, fixed-order tree => determ.
// ===========================================================================
__global__ __launch_bounds__(256) void k_bnstats(
    const float* __restrict__ partial, const float* __restrict__ gamma,
    const float* __restrict__ beta, float* __restrict__ coef,
    int nblk, float invE)
{
  __shared__ float ls[256], lq[256];
  const int c = blockIdx.x, t = threadIdx.x;
  float s = 0.f, q = 0.f;
  for (int b = t; b < nblk; b += 256) {
    s += partial[(size_t)b * 256 + c];
    q += partial[(size_t)b * 256 + 128 + c];
  }
  ls[t] = s; lq[t] = q; __syncthreads();
  for (int st = 128; st > 0; st >>= 1) {
    if (t < st) { ls[t] += ls[t + st]; lq[t] += lq[t + st]; }
    __syncthreads();
  }
  if (t == 0) {
    float mean = ls[0] * invE;
    float var  = lq[0] * invE - mean * mean;       // biased var (jnp.var)
    float rstd = rsqrtf(var + 1e-5f);
    float a = gamma[c] * rstd;
    coef[c] = a;
    coef[128 + c] = beta[c] - mean * a;
  }
}

// ===========================================================================
// Kernel 3: layer-2 GEMM  h2 = relu(bn1(h1)) @ W2^T + b2  (+ stats of h2)
// ===========================================================================
__global__ __launch_bounds__(128) void k_layer2(
    const uint16_t* __restrict__ h1, const float* __restrict__ W2,
    const float* __restrict__ b2, const float* __restrict__ coef,
    uint16_t* __restrict__ h2, float* __restrict__ partial,
    int E, int ntiles)
{
  __shared__ __align__(16) uint32_t sW[128 * 68];          // W2 bf16, pitch 136 half
  __shared__ __align__(16) float    sStage[WAVES][16 * 132];
  __shared__ float sStats[WAVES][256];
  __shared__ __align__(16) float sCoef[256];               // [A | B]
  __shared__ float sBias[128];

  const int tid = threadIdx.x;
  {
    const int row = tid;
    const float4* src = reinterpret_cast<const float4*>(W2 + row * 128);
    uint32_t* drow = &sW[row * 68];
    #pragma unroll
    for (int k4 = 0; k4 < 32; ++k4) {
      float4 w = src[k4];
      drow[k4 * 2 + 0] = pack2(w.x, w.y);
      drow[k4 * 2 + 1] = pack2(w.z, w.w);
    }
    sBias[row] = b2[row];
    sCoef[row] = coef[row];
    sCoef[128 + row] = coef[128 + row];
  }
  __syncthreads();

  const int wave = tid >> 5, lane = tid & 31;
  const int mrow = lane & 15, half = lane >> 4;
  const int gwave  = blockIdx.x * WAVES + wave;
  const int nwaves = gridDim.x * WAVES;
  const int iters  = (ntiles + nwaves - 1) / nwaves;

  float biasr[8];
  #pragma unroll
  for (int nt = 0; nt < 8; ++nt) biasr[nt] = sBias[nt * 16 + mrow];

  float sumA[8], sumQ[8];
  #pragma unroll
  for (int i = 0; i < 8; ++i) { sumA[i] = 0.f; sumQ[i] = 0.f; }

  for (int it = 0; it < iters; ++it) {
    const int tile = gwave + it * nwaves;
    const bool tvalid = tile < ntiles;
    const int e0 = tile * 16;
    const bool fulltile = (e0 + 16 <= E);

    Frag8 af[4];
    if (tvalid) {
      int e = e0 + mrow;
      if (!fulltile && e >= E) e = E - 1;
      const uint4* hp = reinterpret_cast<const uint4*>(h1) + (size_t)e * 16;
      #pragma unroll
      for (int ks = 0; ks < 4; ++ks) {
        #pragma unroll
        for (int j = 0; j < 2; ++j) {
          int c = ks * 32 + half * 16 + j * 8;
          uint4 raw = hp[c >> 3];
          uint32_t rw[4] = {raw.x, raw.y, raw.z, raw.w};
          #pragma unroll
          for (int d = 0; d < 4; ++d) {
            int cc = c + d * 2;
            float f0 = bf2f(rw[d] & 0xffffu);
            float f1 = bf2f(rw[d] >> 16);
            float t0 = fmaxf(fmaf(f0, sCoef[cc],     sCoef[128 + cc]),     0.f);
            float t1 = fmaxf(fmaf(f1, sCoef[cc + 1], sCoef[128 + cc + 1]), 0.f);
            af[ks].u[j * 4 + d] = pack2(t0, t1);
          }
        }
      }
      #pragma unroll
      for (int nt = 0; nt < 8; ++nt) {
        const int brow = nt * 16 + mrow;
        const uint32_t* wbase = &sW[brow * 68 + half * 8];
        Frag8 bfr[4];
        #pragma unroll
        for (int ks = 0; ks < 4; ++ks) {
          const uint4* bp = reinterpret_cast<const uint4*>(wbase + ks * 16);
          uint4 q0 = bp[0], q1 = bp[1];
          bfr[ks].u[0] = q0.x; bfr[ks].u[1] = q0.y;
          bfr[ks].u[2] = q0.z; bfr[ks].u[3] = q0.w;
          bfr[ks].u[4] = q1.x; bfr[ks].u[5] = q1.y;
          bfr[ks].u[6] = q1.z; bfr[ks].u[7] = q1.w;
        }
        v8f acc;
        #pragma unroll
        for (int i = 0; i < 8; ++i) acc[i] = biasr[nt];
        #pragma unroll
        for (int ks = 0; ks < 4; ++ks) {
          acc = __builtin_amdgcn_wmma_f32_16x16x32_bf16(
              false, __builtin_bit_cast(v16bf, af[ks]),
              false, __builtin_bit_cast(v16bf, bfr[ks]),
              (short)0, acc, false, false);
        }
        const int n = nt * 16 + mrow;
        if (fulltile) {
          #pragma unroll
          for (int r = 0; r < 8; ++r) {
            float v = acc[r];
            sumA[nt] += v; sumQ[nt] += v * v;
            sStage[wave][(r + half * 8) * 132 + n] = v;
          }
        } else {
          #pragma unroll
          for (int r = 0; r < 8; ++r) {
            float v = acc[r];
            int mr = r + half * 8;
            if (e0 + mr < E) { sumA[nt] += v; sumQ[nt] += v * v; }
            sStage[wave][mr * 132 + n] = v;
          }
        }
      }
    }
    __syncthreads();
    if (tvalid) {
      #pragma unroll
      for (int i = 0; i < 8; ++i) {
        int row  = i * 2 + half;
        int colf = mrow * 8;
        if (fulltile || (e0 + row < E)) {
          const float* sp = &sStage[wave][row * 132 + colf];
          float4 f0 = *reinterpret_cast<const float4*>(sp);
          float4 f1 = *reinterpret_cast<const float4*>(sp + 4);
          uint4 o;
          o.x = pack2(f0.x, f0.y); o.y = pack2(f0.z, f0.w);
          o.z = pack2(f1.x, f1.y); o.w = pack2(f1.z, f1.w);
          size_t off = (size_t)(e0 + row) * 128 + colf;
          reinterpret_cast<uint4*>(h2)[off >> 3] = o;
        }
      }
    }
    __syncthreads();
  }

  #pragma unroll
  for (int nt = 0; nt < 8; ++nt) {
    float a = sumA[nt] + __shfl_xor(sumA[nt], 16);
    float q = sumQ[nt] + __shfl_xor(sumQ[nt], 16);
    if (half == 0) {
      sStats[wave][nt * 16 + mrow]       = a;
      sStats[wave][128 + nt * 16 + mrow] = q;
    }
  }
  __syncthreads();
  {
    const int t = tid;
    float s = sStats[0][t] + sStats[1][t] + sStats[2][t] + sStats[3][t];
    float q = sStats[0][128 + t] + sStats[1][128 + t] +
              sStats[2][128 + t] + sStats[3][128 + t];
    partial[(size_t)blockIdx.x * 256 + t]       = s;
    partial[(size_t)blockIdx.x * 256 + 128 + t] = q;
  }
}

// ===========================================================================
// Kernel 5: layer-3 GEMV  out[e] = relu(bn2(h2[e])) . w3 + b3
// ===========================================================================
__global__ __launch_bounds__(256) void k_layer3(
    const uint16_t* __restrict__ h2, const float* __restrict__ coef,
    const float* __restrict__ W3, const float* __restrict__ b3,
    float* __restrict__ out, int E)
{
  __shared__ float sw3[128], sca[128], scb[128];
  const int t = threadIdx.x;
  if (t < 128) { sw3[t] = W3[t]; sca[t] = coef[t]; scb[t] = coef[128 + t]; }
  __syncthreads();
  const int e = blockIdx.x * 256 + t;
  if (e >= E) return;
  const uint4* hp = reinterpret_cast<const uint4*>(h2) + (size_t)e * 16;
  float acc = b3[0];
  #pragma unroll
  for (int kc = 0; kc < 16; ++kc) {
    uint4 raw = hp[kc];
    uint32_t rw[4] = {raw.x, raw.y, raw.z, raw.w};
    #pragma unroll
    for (int d = 0; d < 4; ++d) {
      int c = kc * 8 + d * 2;
      float f0 = bf2f(rw[d] & 0xffffu);
      float f1 = bf2f(rw[d] >> 16);
      float t0 = fmaxf(fmaf(f0, sca[c],     scb[c]),     0.f);
      float t1 = fmaxf(fmaf(f1, sca[c + 1], scb[c + 1]), 0.f);
      acc = fmaf(t0, sw3[c], acc);
      acc = fmaf(t1, sw3[c + 1], acc);
    }
  }
  out[e] = acc;
}

// ===========================================================================
extern "C" void kernel_launch(void* const* d_in, const int* in_sizes, int n_in,
                              void* d_out, int out_size, void* d_ws, size_t ws_size,
                              hipStream_t stream)
{
  const float* x   = (const float*)d_in[0];
  const int*   ei  = (const int*)d_in[1];
  const float* W1  = (const float*)d_in[2];
  const float* b1  = (const float*)d_in[3];
  const float* g1  = (const float*)d_in[4];
  const float* be1 = (const float*)d_in[5];
  const float* W2  = (const float*)d_in[6];
  const float* b2  = (const float*)d_in[7];
  const float* g2  = (const float*)d_in[8];
  const float* be2 = (const float*)d_in[9];
  const float* W3  = (const float*)d_in[10];
  const float* b3  = (const float*)d_in[11];
  float* out = (float*)d_out;

  const int E = in_sizes[1] / 2;
  const int ntiles = (E + 15) / 16;

  char* ws = (char*)d_ws;
  size_t off = 0;
  auto alloc = [&](size_t bytes) -> void* {
    void* p = ws + off;
    off += bytes;
    off = (off + 255) & ~(size_t)255;
    return p;
  };
  uint16_t* h1   = (uint16_t*)alloc((size_t)E * 128 * 2);
  uint16_t* h2   = (uint16_t*)alloc((size_t)E * 128 * 2);
  float* part1   = (float*)alloc((size_t)NBLK * 256 * 4);
  float* part2   = (float*)alloc((size_t)NBLK * 256 * 4);
  float* coef1   = (float*)alloc(256 * 4);
  float* coef2   = (float*)alloc(256 * 4);
  (void)ws_size; (void)n_in; (void)out_size;

  const float invE = 1.0f / (float)E;

  k_layer1 <<<NBLK, 128, 0, stream>>>(x, ei, W1, b1, h1, part1, E, ntiles);
  k_bnstats<<<128, 256, 0, stream>>>(part1, g1, be1, coef1, NBLK, invE);
  k_layer2 <<<NBLK, 128, 0, stream>>>(h1, W2, b2, coef1, h2, part2, E, ntiles);
  k_bnstats<<<128, 256, 0, stream>>>(part2, g2, be2, coef2, NBLK, invE);
  k_layer3 <<<(E + 255) / 256, 256, 0, stream>>>(h2, coef2, W3, b3, out, E);
}
